// GNNModel_41154376630861
// MI455X (gfx1250) — matrix operand order repaired and checked by reading, not compiled, not run on hardware
//
#include <hip/hip_runtime.h>
#include <hip/hip_bf16.h>

typedef __attribute__((ext_vector_type(16))) __bf16 v16bf;
typedef __attribute__((ext_vector_type(8)))  float  v8f;

// ---------------------------------------------------------------------------
// Weight pre-pack: produce the exact per-lane v16bf each wave lane consumes.
// B-matrix 32x16 bf16, contiguous k order (same as A): lanes 0-15 hold K=0..15 (col=lane),
// lanes 16-31 hold K=16..31 (col=lane-16), 2 values per VGPR.
// Packed index: (((kt*(M/16) + ctg)*32 + lane)*16 + i)
// ---------------------------------------------------------------------------
__global__ void pack_w_kernel(const float* __restrict__ W, __bf16* __restrict__ Wp,
                              int K, int M) {
  int idx = blockIdx.x * blockDim.x + threadIdx.x;
  int total = (K / 32) * (M / 16) * 32;
  if (idx >= total) return;
  int lane = idx & 31;
  int t    = idx >> 5;
  int mt   = M >> 4;
  int ctg  = t % mt;
  int kt   = t / mt;
  int col  = ctg * 16 + (lane & 15);
  int kbase = kt * 32 + (lane >> 4) * 16;
  __bf16* out = Wp + (size_t)idx * 16;
#pragma unroll
  for (int i = 0; i < 16; ++i)
    out[i] = (__bf16)W[(size_t)(kbase + i) * M + col];
}

// ---------------------------------------------------------------------------
// WMMA GEMM: Y[N,M] = epilogue(X[N,K] @ W[K,M])
//   - one wave computes a 16x64 output tile (4 x v_wmma_f32_16x16x32_bf16)
//   - X loaded fp32 -> bf16 on the fly in contiguous k order (as packed W); edge rows are
//     clamped (never stored) so the K-loop is branch-free with full EXEC
//   - W read from pre-packed bf16 layout: running pointer + immediate offsets
//   - optional bias, relu, residual-add, and row-gather (rowidx)
// ---------------------------------------------------------------------------
__global__ __launch_bounds__(256)
void gemm_wmma_kernel(const float* __restrict__ X, const __bf16* __restrict__ Wp,
                      const float* __restrict__ bias, const float* __restrict__ resid,
                      const int* __restrict__ rowidx, float* __restrict__ Y,
                      int N, int K, int M, int relu_flag) {
  const int wave  = threadIdx.x >> 5;
  const int lane  = threadIdx.x & 31;
  const int rt    = blockIdx.x * 8 + wave;   // 16-row tile index (uniform per wave)
  const int row0  = rt * 16;
  if (row0 >= N) return;                      // uniform per wave -> EXEC stays full
  const int col0  = blockIdx.y * 64;
  const int lrow  = lane & 15;
  const int khalf = lane >> 4;
  int r = row0 + lrow;
  if (r >= N) r = N - 1;                      // clamp: garbage rows are never stored
  const int xr = rowidx ? rowidx[r] : r;

  v8f acc[4];
#pragma unroll
  for (int ct = 0; ct < 4; ++ct)
#pragma unroll
    for (int v = 0; v < 8; ++v) acc[ct][v] = 0.f;

  const int mt     = M >> 4;
  const int ktiles = K >> 5;
  const int ctg0   = blockIdx.y * 4;

  // running pointers: A advances 32 floats/kt, B advances one kt-slab (mt*512 elems)
  const float*  ap = X + (size_t)xr * K + khalf * 16;
  const __bf16* bp = Wp + ((size_t)ctg0 * 32 + lane) * 16;
  const size_t  kstride = (size_t)mt * 512;   // bf16 elems per kt step (mt tiles * 32 lanes * 16)

  for (int kt = 0; kt < ktiles; ++kt) {
    // A-matrix 16x32 bf16 layout: lane-half selects contiguous 16-chunk:
    //   elems 0..15 = K[kt*32 + khalf*16 + 0..15]
    const float4 c00 = *(const float4*)(ap);
    const float4 c01 = *(const float4*)(ap + 4);
    const float4 c10 = *(const float4*)(ap + 8);
    const float4 c11 = *(const float4*)(ap + 12);
    // B tiles for this kt: 4 contiguous 1KB tiles, immediate offsets
    const v16bf b0 = *(const v16bf*)(bp);
    const v16bf b1 = *(const v16bf*)(bp + 512);
    const v16bf b2 = *(const v16bf*)(bp + 1024);
    const v16bf b3 = *(const v16bf*)(bp + 1536);

    v16bf a;
    a[0]  = (__bf16)c00.x; a[1]  = (__bf16)c00.y; a[2]  = (__bf16)c00.z; a[3]  = (__bf16)c00.w;
    a[4]  = (__bf16)c01.x; a[5]  = (__bf16)c01.y; a[6]  = (__bf16)c01.z; a[7]  = (__bf16)c01.w;
    a[8]  = (__bf16)c10.x; a[9]  = (__bf16)c10.y; a[10] = (__bf16)c10.z; a[11] = (__bf16)c10.w;
    a[12] = (__bf16)c11.x; a[13] = (__bf16)c11.y; a[14] = (__bf16)c11.z; a[15] = (__bf16)c11.w;

    acc[0] = __builtin_amdgcn_wmma_f32_16x16x32_bf16(false, a, false, b0, (short)0, acc[0], false, false);
    acc[1] = __builtin_amdgcn_wmma_f32_16x16x32_bf16(false, a, false, b1, (short)0, acc[1], false, false);
    acc[2] = __builtin_amdgcn_wmma_f32_16x16x32_bf16(false, a, false, b2, (short)0, acc[2], false, false);
    acc[3] = __builtin_amdgcn_wmma_f32_16x16x32_bf16(false, a, false, b3, (short)0, acc[3], false, false);

    ap += 32;
    bp += kstride;
  }

  // C/D layout: VGPR v -> row v + 8*lanehalf, col = lane&15 within tile
#pragma unroll
  for (int ct = 0; ct < 4; ++ct) {
    const int c  = col0 + ct * 16 + lrow;
    const float bv = bias ? bias[c] : 0.f;
#pragma unroll
    for (int v = 0; v < 8; ++v) {
      const int rr = row0 + v + 8 * khalf;
      if (rr < N) {
        float val = acc[ct][v] + bv;
        if (relu_flag) val = val > 0.f ? val : 0.f;
        if (resid) val += resid[(size_t)rr * M + c];
        Y[(size_t)rr * M + c] = val;
      }
    }
  }
}

// ---------------------------------------------------------------------------
// Edge scatter-add (message passing):  Y[dst] += f(X[src])
//   blockDim.x == F (feature dim) -> no integer division anywhere;
//   src/dst loads are block-uniform. Optional pe3 add and GCN sym-norm scale.
// ---------------------------------------------------------------------------
__global__ void scatter_edges_kernel(const float* __restrict__ X,
                                     const int* __restrict__ src,
                                     const int* __restrict__ dst,
                                     float* __restrict__ Y, long long E,
                                     const float* __restrict__ pe,
                                     const float* __restrict__ dinv) {
  const int F = blockDim.x;
  const int f = threadIdx.x;
  for (long long e = blockIdx.x; e < E; e += gridDim.x) {
    int s = src[e], d = dst[e];
    float v = X[(size_t)s * F + f];
    if (pe)   v += pe[(size_t)s * 64 + (f & 63)];
    if (dinv) v *= dinv[s] * dinv[d];
    atomicAdd(&Y[(size_t)d * F + f], v);
  }
}

__global__ void deg_kernel(const int* __restrict__ dst, float* __restrict__ deg, long long E) {
  for (long long e = blockIdx.x * (long long)blockDim.x + threadIdx.x; e < E;
       e += (long long)gridDim.x * blockDim.x)
    atomicAdd(&deg[dst[e]], 1.f);
}

__global__ void dinv_kernel(float* __restrict__ deg, int N) {
  int i = blockIdx.x * blockDim.x + threadIdx.x;
  if (i < N) deg[i] = rsqrtf(deg[i] + 1.f);
}

// AGG = relu(AGG + H*dinv(n)^2 + bias)   (GCN self-loop + bias + relu)
// blockDim.x == F, grid-stride over nodes.
__global__ void gcn_finish_kernel(float* __restrict__ AGG, const float* __restrict__ H,
                                  const float* __restrict__ bias,
                                  const float* __restrict__ dinv, int N) {
  const int F = blockDim.x;
  const int f = threadIdx.x;
  const float bv = bias[f];
  for (int n = blockIdx.x; n < N; n += gridDim.x) {
    float di = dinv[n];
    size_t i = (size_t)n * F + f;
    float v = AGG[i] + H[i] * di * di + bv;
    AGG[i] = v > 0.f ? v : 0.f;
  }
}

__global__ void add3_kernel(const float* __restrict__ a, const float* __restrict__ b,
                            const float* __restrict__ c, float* __restrict__ y, long long n) {
  for (long long i = blockIdx.x * (long long)blockDim.x + threadIdx.x; i < n;
       i += (long long)gridDim.x * blockDim.x)
    y[i] = a[i] + b[i] + c[i];
}

__global__ void fill_kernel(float* __restrict__ p, long long n, float v) {
  for (long long i = blockIdx.x * (long long)blockDim.x + threadIdx.x; i < n;
       i += (long long)gridDim.x * blockDim.x)
    p[i] = v;
}

// ----------------------- segment softmax (attention) -----------------------
__device__ inline void atomicMaxF(float* addr, float val) {
  unsigned int* ai = (unsigned int*)addr;
  unsigned int old = __float_as_uint(*addr);
  while (__uint_as_float(old) < val) {
    unsigned int assumed = old;
    old = atomicCAS(ai, assumed, __float_as_uint(val));
    if (old == assumed) break;
  }
}

// blockDim.x == F (=192); grid-stride over nodes; batch[n] is block-uniform.
__global__ void seg_max_kernel(const float* __restrict__ gate, const int* __restrict__ batch,
                               float* __restrict__ gmax, int N) {
  const int F = blockDim.x;
  const int f = threadIdx.x;
  for (int n = blockIdx.x; n < N; n += gridDim.x)
    atomicMaxF(&gmax[(size_t)batch[n] * F + f], gate[(size_t)n * F + f]);
}

__global__ void seg_esum_kernel(const float* __restrict__ gate, const int* __restrict__ batch,
                                const float* __restrict__ gmax, float* __restrict__ esum,
                                int N) {
  const int F = blockDim.x;
  const int f = threadIdx.x;
  for (int n = blockIdx.x; n < N; n += gridDim.x) {
    int g = batch[n];
    atomicAdd(&esum[(size_t)g * F + f],
              expf(gate[(size_t)n * F + f] - gmax[(size_t)g * F + f]));
  }
}

__global__ void seg_zg_kernel(const float* __restrict__ gate, const float* __restrict__ z,
                              const int* __restrict__ batch, const float* __restrict__ gmax,
                              const float* __restrict__ esum, float* __restrict__ zg,
                              int N) {
  const int F = blockDim.x;
  const int f = threadIdx.x;
  for (int n = blockIdx.x; n < N; n += gridDim.x) {
    int g = batch[n];
    size_t gi = (size_t)g * F + f;
    float a = expf(gate[(size_t)n * F + f] - gmax[gi]) / esum[gi];
    atomicAdd(&zg[gi], a * z[(size_t)n * F + f]);
  }
}

// ---------------------------------------------------------------------------
// Orchestration
// ---------------------------------------------------------------------------
static inline int cap_blocks(long long want) {
  if (want > 65536) want = 65536;
  if (want < 1) want = 1;
  return (int)want;
}

extern "C" void kernel_launch(void* const* d_in, const int* in_sizes, int n_in,
                              void* d_out, int out_size, void* d_ws, size_t ws_size,
                              hipStream_t stream) {
  (void)n_in; (void)out_size; (void)ws_size;
  const float* x1   = (const float*)d_in[0];
  const float* x2   = (const float*)d_in[1];
  const float* pe   = (const float*)d_in[2];
  const int*   eidx = (const int*)d_in[3];
  const int*   batch= (const int*)d_in[4];
  const float* Wc  = (const float*)d_in[5];  const float* bc  = (const float*)d_in[6];
  const float* Wmf = (const float*)d_in[7];  const float* bmf = (const float*)d_in[8];
  const float* Wgf = (const float*)d_in[9];  const float* bgf = (const float*)d_in[10];
  const float* Wmc = (const float*)d_in[11]; const float* bmc = (const float*)d_in[12];
  const float* Wgc = (const float*)d_in[13]; const float* bgc = (const float*)d_in[14];
  const float* Wa1 = (const float*)d_in[15]; const float* ba1 = (const float*)d_in[16];
  const float* Wa2 = (const float*)d_in[17]; const float* ba2 = (const float*)d_in[18];
  const float* Wgd = (const float*)d_in[19]; const float* bgd = (const float*)d_in[20];
  const float* Wd1 = (const float*)d_in[21]; const float* bd1 = (const float*)d_in[22];
  const float* Wd2 = (const float*)d_in[23]; const float* bd2 = (const float*)d_in[24];
  const float* Wd3 = (const float*)d_in[25]; const float* bd3 = (const float*)d_in[26];

  const int       N = in_sizes[0] / 384;
  const long long E = in_sizes[3] / 2;
  const int* src = eidx;
  const int* dst = eidx + E;
  const int  G   = 64;  // NUM_GRAPHS

  // ---- workspace carve (256B aligned) ----
  char* wsb = (char*)d_ws;
  size_t off = 0;
  auto carve = [&](size_t bytes) -> char* {
    char* p = wsb + off;
    off += (bytes + 255) & ~(size_t)255;
    return p;
  };
  float* R    = (float*)carve((size_t)N * 192 * 4);   // x1_res
  float* A    = (float*)carve((size_t)N * 192 * 4);   // scratch
  float* B    = (float*)carve((size_t)N * 192 * 4);   // scratch
  float* D    = (float*)carve((size_t)N * 192 * 4);   // scratch
  float* dinv = (float*)carve((size_t)N * 4);
  float* gmax = (float*)carve((size_t)G * 192 * 4);
  float* esum = (float*)carve((size_t)G * 192 * 4);
  float* zg   = (float*)carve((size_t)G * 192 * 4);
  __bf16* WPbase = (__bf16*)carve((size_t)468992 * 2);
  float* C = (float*)d_out;   // reuse output buffer (N*384) as 256-wide scratch

  // ---- pack all weights to WMMA-ready bf16 ----
  size_t wo = 0;
  auto wp_next = [&](int K, int M) -> __bf16* {
    __bf16* p = WPbase + wo;
    wo += (size_t)K * M;
    return p;
  };
  auto pack = [&](const float* W, __bf16* dstp, int K, int M) {
    int tot = (K / 32) * (M / 16) * 32;
    pack_w_kernel<<<(tot + 255) / 256, 256, 0, stream>>>(W, dstp, K, M);
  };
  __bf16 *pWc  = wp_next(384, 192), *pWmf = wp_next(192, 256), *pWgf = wp_next(256, 192);
  __bf16 *pWmc = wp_next(32, 64),   *pWgc = wp_next(64, 192);
  __bf16 *pWa1 = wp_next(192, 256), *pWa2 = wp_next(256, 192);
  __bf16 *pWgd = wp_next(192, 192), *pWd1 = wp_next(192, 192);
  __bf16 *pWd2 = wp_next(192, 192), *pWd3 = wp_next(192, 384);
  pack(Wc, pWc, 384, 192);  pack(Wmf, pWmf, 192, 256); pack(Wgf, pWgf, 256, 192);
  pack(Wmc, pWmc, 32, 64);  pack(Wgc, pWgc, 64, 192);
  pack(Wa1, pWa1, 192, 256); pack(Wa2, pWa2, 256, 192);
  pack(Wgd, pWgd, 192, 192); pack(Wd1, pWd1, 192, 192);
  pack(Wd2, pWd2, 192, 192); pack(Wd3, pWd3, 192, 384);

  auto gemm = [&](const float* X, const __bf16* Wp, const float* bias, const float* resid,
                  const int* rowidx, float* Y, int K, int M, bool relu) {
    int row_tiles = (N + 15) / 16;
    dim3 grid((row_tiles + 7) / 8, M / 64);
    gemm_wmma_kernel<<<grid, 256, 0, stream>>>(X, Wp, bias, resid, rowidx, Y, N, K, M,
                                               relu ? 1 : 0);
  };
  auto scatter = [&](const float* X, float* Y, int F, const float* peP, const float* dinvP) {
    scatter_edges_kernel<<<cap_blocks(E), F, 0, stream>>>(X, src, dst, Y, E, peP, dinvP);
  };
  auto zero = [&](float* p, size_t floats) { hipMemsetAsync(p, 0, floats * 4, stream); };
  auto gcnfin = [&](float* AGG, const float* H, const float* bias) {
    gcn_finish_kernel<<<cap_blocks(N), 192, 0, stream>>>(AGG, H, bias, dinv, N);
  };

  // ========================== encoder ==========================
  // 1) x1_res = relu(x1 @ Wc + bc)
  gemm(x1, pWc, bc, nullptr, nullptr, R, 384, 192, true);
  // 2) agg = scatter_add((x1_res + pe3)[src] -> dst)
  zero(B, (size_t)N * 192);
  scatter(R, B, 192, pe, nullptr);
  // 3) h256 = relu(agg @ Wmf + bmf)
  gemm(B, pWmf, bmf, nullptr, nullptr, C, 192, 256, true);
  // 4) hW = h256 @ Wgf  (bias deferred to gcn_finish)
  gemm(C, pWgf, nullptr, nullptr, nullptr, A, 256, 192, false);
  // 5) degree -> dinv = rsqrt(deg+1)
  zero(dinv, (size_t)N);
  deg_kernel<<<cap_blocks((E + 255) / 256), 256, 0, stream>>>(dst, dinv, E);
  dinv_kernel<<<(N + 255) / 256, 256, 0, stream>>>(dinv, N);
  // 6) GCN aggregate with sym-norm
  zero(B, (size_t)N * 192);
  scatter(A, B, 192, nullptr, dinv);
  // 7) h = relu(agg + hW*dinv^2 + bgf)   -> B
  gcnfin(B, A, bgf);
  // 8) class path: agg32 = scatter_add(x2[src])
  zero(A, (size_t)N * 32);
  scatter(x2, A, 32, nullptr, nullptr);
  // 9) c64 = relu(agg32 @ Wmc + bmc)
  gemm(A, pWmc, bmc, nullptr, nullptr, C, 32, 64, true);
  // 10) cW = c64 @ Wgc
  gemm(C, pWgc, nullptr, nullptr, nullptr, A, 64, 192, false);
  // 11) GCN class: agg + self-loop + bias + relu -> D
  zero(D, (size_t)N * 192);
  scatter(A, D, 192, nullptr, dinv);
  gcnfin(D, A, bgc);
  // 12) z = h + c + x1_res -> A
  add3_kernel<<<cap_blocks(((long long)N * 192 + 255) / 256), 256, 0, stream>>>(
      B, D, R, A, (long long)N * 192);

  // =================== attentional aggregation =================
  // 13) gate = relu(z @ Wa1 + ba1) @ Wa2 + ba2
  gemm(A, pWa1, ba1, nullptr, nullptr, C, 192, 256, true);
  gemm(C, pWa2, ba2, nullptr, nullptr, B, 256, 192, false);
  // 14) segment softmax + weighted sum
  fill_kernel<<<cap_blocks(((long long)G * 192 + 255) / 256), 256, 0, stream>>>(
      gmax, (long long)G * 192, -3.0e38f);
  seg_max_kernel<<<cap_blocks(N), 192, 0, stream>>>(B, batch, gmax, N);
  zero(esum, (size_t)G * 192);
  seg_esum_kernel<<<cap_blocks(N), 192, 0, stream>>>(B, batch, gmax, esum, N);
  zero(zg, (size_t)G * 192);
  seg_zg_kernel<<<cap_blocks(N), 192, 0, stream>>>(B, A, batch, gmax, esum, zg, N);

  // ========================== decoder ==========================
  // 15) z1W = zg[batch] @ Wgd   (row-gather GEMM)
  gemm(zg, pWgd, nullptr, nullptr, batch, B, 192, 192, false);
  // 16) z1_res = relu(GCN(z1)) -> D
  zero(D, (size_t)N * 192);
  scatter(B, D, 192, nullptr, dinv);
  gcnfin(D, B, bgd);
  // 17) t = mpnn(z1_res + pe3): scatter then GEMM
  zero(A, (size_t)N * 192);
  scatter(D, A, 192, pe, nullptr);
  gemm(A, pWd1, bd1, nullptr, nullptr, B, 192, 192, true);   // t1
  // 18) t2 = relu(mpnn(t1)) + z1_res
  zero(A, (size_t)N * 192);
  scatter(B, A, 192, nullptr, nullptr);
  gemm(A, pWd2, bd2, D, nullptr, B, 192, 192, true);         // resid = z1_res
  // 19) out = relu(mpnn(t2) @ Wd3 + bd3)
  zero(A, (size_t)N * 192);
  scatter(B, A, 192, nullptr, nullptr);
  gemm(A, pWd3, bd3, nullptr, nullptr, (float*)d_out, 192, 384, true);
}
